// AttentionCropBlock_7859790151794
// MI455X (gfx1250) — matrix operand order repaired and checked by reading, not compiled, not run on hardware
//
#include <hip/hip_runtime.h>
#include <hip/hip_fp16.h>

// ---------------------------------------------------------------------------
// AttentionCropBlock for MI455X (gfx1250, wave32, WMMA + TDM)
//
//   k1: conv1 3->64 3x3 + BN + ReLU      -> h1 (fp16)
//   k2: conv2 64->64 3x3 + BN + ReLU     -> h2 (fp16)   [WMMA, 9 shifted GEMMs]
//   k3: conv3 64->64 3x3 + BN + fused channel avg/max   [WMMA, 9 shifted GEMMs]
//   k4: 7x7 (2->1) conv + sigmoid        -> sa
//   k5: horizontal 256-window sums       -> T           [TDM tensor_load_to_lds]
//   k6: vertical sliding sums + argmax   -> rowcol
//   k7: gather crop of x                 -> out (8,3,256,256) f32
//
// conv2/conv3: 3x3 conv = 9 shifted 1x1 GEMMs (K=64). LDS layouts put Cin
// innermost and pad 64->72 halves (144B row stride) so every WMMA fragment is
// two 16B ds_load_b128 covering all 64 banks conflict-free.
// ---------------------------------------------------------------------------

typedef __attribute__((ext_vector_type(16))) _Float16     v16h;
typedef __attribute__((ext_vector_type(8)))  _Float16     v8h;
typedef __attribute__((ext_vector_type(8)))  float        v8f;
typedef __attribute__((ext_vector_type(4)))  unsigned int u32x4;
typedef __attribute__((ext_vector_type(8)))  unsigned int u32x8;

#define Hh      512
#define Ww      512
#define Bb      8
#define C1      64
#define KCROP   256
#define EPSf    1e-5f
#define TPX     128          // pixels (one row segment) per block
#define CPAD    72           // padded Cin stride in halves (144B rows)

// ---------------------------------------------------------------- conv1 ----
__global__ void conv1_bn_relu(const float* __restrict__ x, const float* __restrict__ w1,
                              const float* __restrict__ g, const float* __restrict__ bt,
                              const float* __restrict__ mn, const float* __restrict__ vr,
                              _Float16* __restrict__ h1) {
    const int tid = threadIdx.x;
    const int xx  = blockIdx.x * 256 + tid;
    const int y   = blockIdx.y;
    const int z   = blockIdx.z;          // b*64 + co
    const int co  = z & 63;
    const int b   = z >> 6;

    float wreg[27];
#pragma unroll
    for (int i = 0; i < 27; ++i) wreg[i] = w1[co * 27 + i];   // uniform -> scalar

    const float scale = g[co] * rsqrtf(vr[co] + EPSf);
    const float bias  = bt[co] - mn[co] * scale;

    float acc = 0.f;
#pragma unroll
    for (int ci = 0; ci < 3; ++ci)
#pragma unroll
        for (int ky = 0; ky < 3; ++ky) {
            const int ys = y + ky - 1;
            if (ys < 0 || ys >= Hh) continue;
#pragma unroll
            for (int kx = 0; kx < 3; ++kx) {
                const int xs = xx + kx - 1;
                if (xs < 0 || xs >= Ww) continue;
                acc += wreg[ci * 9 + ky * 3 + kx] *
                       x[(((size_t)b * 3 + ci) * Hh + ys) * Ww + xs];
            }
        }
    const float r = fmaxf(acc * scale + bias, 0.f);
    h1[(((size_t)b * C1 + co) * Hh + y) * Ww + xx] = (_Float16)r;
}

// -------------------------------------------------- conv2 / conv3 (WMMA) ---
// Block = 256 threads (8 waves). Tile: 64 couts x 128 pixels of one row.
// wave wv: cout tile = (wv>>1)*16, pixel group = (wv&1)*64 (4 subtiles of 16).
template <bool FUSE_STATS>
__global__ void conv_wmma(const _Float16* __restrict__ src, const float* __restrict__ w,
                          const float* __restrict__ g, const float* __restrict__ bt,
                          const float* __restrict__ mn, const float* __restrict__ vr,
                          _Float16* __restrict__ dst,
                          float* __restrict__ avgp, float* __restrict__ mxp) {
    __shared__ __align__(16) _Float16 lW[9][C1][CPAD];        // 83KB  weights
    __shared__ __align__(16) _Float16 lP[3][TPX + 2][CPAD];   // 56KB  patch
    __shared__ float lScale[C1], lBias[C1];

    const int tid = threadIdx.x;
    const int bid = blockIdx.x;
    const int x0  = (bid & 3) * TPX;
    const int y   = (bid >> 2) & (Hh - 1);
    const int b   =  bid >> 11;            // bid / (4*512)

    __builtin_prefetch(&w[tid * 16], 0, 0);   // global_prefetch_b8

    // Folded BN coefficients.
    if (tid < C1) {
        const float s = g[tid] * rsqrtf(vr[tid] + EPSf);
        lScale[tid] = s;
        lBias[tid]  = bt[tid] - mn[tid] * s;
    }

    // Stage weights f32 -> f16:  lW[r][co][ci] = w[co*576 + ci*9 + r].
    for (int e = tid; e < 9 * C1 * C1; e += 256) {
        const int r  = e >> 12;
        const int co = (e >> 6) & 63;
        const int ci = e & 63;
        lW[r][co][ci] = (_Float16)w[co * 576 + ci * 9 + r];
    }

    // Stage haloed input rows (y-1..y+1, x0-1..x0+128), Cin innermost.
    for (int e = tid; e < 3 * (TPX + 2) * C1; e += 256) {
        const int ci = e & 63;
        const int t  = e >> 6;
        const int xxi = t % (TPX + 2);
        const int dy  = t / (TPX + 2);
        const int ys  = y + dy - 1;
        const int xs  = x0 + xxi - 1;
        _Float16 val = (_Float16)0.f;
        if (ys >= 0 && ys < Hh && xs >= 0 && xs < Ww)
            val = src[(((size_t)b * C1 + ci) * Hh + ys) * Ww + xs];
        lP[dy][xxi][ci] = val;
    }
    __syncthreads();

    const int lane = tid & 31;
    const int wv   = tid >> 5;
    const int co0  = (wv >> 1) << 4;   // 0/16/32/48
    const int pg   = (wv & 1) * 64;    // pixel group base
    const int half = lane >> 4;
    const int m    = lane & 15;        // A row (M) == B/C column (N)

    v8f acc[4] = {v8f{}, v8f{}, v8f{}, v8f{}};

#pragma unroll
    for (int r = 0; r < 9; ++r) {                 // shift (ky,kx)
        const int ky = r / 3, kx = r % 3;
#pragma unroll
        for (int kc = 0; kc < 2; ++kc) {          // Cin halves of 32
            const int k0 = kc * 32;
            // A 16x32 f16: two contiguous 8-half runs per lane.
            v16h a;
            *((v8h*)&a)       = *(const v8h*)&lW[r][co0 + m][k0 + half * 8];
            *(((v8h*)&a) + 1) = *(const v8h*)&lW[r][co0 + m][k0 + 16 + half * 8];
#pragma unroll
            for (int st = 0; st < 4; ++st) {      // 4 pixel subtiles, reuse A
                const int xxi = pg + st * 16 + m + kx;   // patch x index
                // B 32x16 f16: two contiguous 8-half runs per lane (same K order as A).
                v16h bm;
                *((v8h*)&bm)       = *(const v8h*)&lP[ky][xxi][k0 + half * 8];
                *(((v8h*)&bm) + 1) = *(const v8h*)&lP[ky][xxi][k0 + 16 + half * 8];
                acc[st] = __builtin_amdgcn_wmma_f32_16x16x32_f16(
                    false, a, false, bm, (short)0, acc[st], false, false);
            }
        }
    }

    // Epilogue.  C/D layout: M = r + 8*half (cout row), N = lane%16 (pixel).
    if constexpr (!FUSE_STATS) {
#pragma unroll
        for (int st = 0; st < 4; ++st) {
            const int px = pg + st * 16 + m;
#pragma unroll
            for (int r = 0; r < 8; ++r) {
                const int co = co0 + half * 8 + r;
                const float vv = fmaxf(acc[st][r] * lScale[co] + lBias[co], 0.f);
                dst[(((size_t)b * C1 + co) * Hh + y) * Ww + (x0 + px)] = (_Float16)vv;
            }
        }
    } else {
        // Reuse the (now dead) weight LDS for the 64x128 f32 feature tile.
        float* lFeat = (float*)&lW[0][0][0];                // 32KB of 83KB
        __syncthreads();                                    // lW reads done
#pragma unroll
        for (int st = 0; st < 4; ++st) {
            const int px = pg + st * 16 + m;
#pragma unroll
            for (int r = 0; r < 8; ++r) {
                const int co = co0 + half * 8 + r;
                lFeat[co * TPX + px] = acc[st][r] * lScale[co] + lBias[co];
            }
        }
        __syncthreads();
        if (tid < TPX) {   // channel avg / max, feat never touches HBM
            float s = 0.f, mxv = -3.0e38f;
#pragma unroll 8
            for (int co = 0; co < C1; ++co) {
                const float f = lFeat[co * TPX + tid];
                s  += f;
                mxv = fmaxf(mxv, f);
            }
            const size_t o = ((size_t)b * Hh + y) * Ww + x0 + tid;
            avgp[o] = s * (1.f / 64.f);
            mxp[o]  = mxv;
        }
    }
}

// ------------------------------------------------ spatial attention conv ---
__global__ void sa_conv_sigmoid(const float* __restrict__ avgp, const float* __restrict__ mxp,
                                const float* __restrict__ wsa, float* __restrict__ sa) {
    __shared__ float lw[2][7][7];
    if (threadIdx.x < 98) ((float*)lw)[threadIdx.x] = wsa[threadIdx.x];
    __syncthreads();

    const int idx = blockIdx.x * 256 + threadIdx.x;
    const int xx  = idx % Ww;
    const int y   = idx / Ww;
    const int b   = blockIdx.y;

    float s = 0.f;
#pragma unroll
    for (int ky = 0; ky < 7; ++ky) {
        const int ys = y + ky - 3;
        if (ys < 0 || ys >= Hh) continue;
#pragma unroll
        for (int kx = 0; kx < 7; ++kx) {
            const int xs = xx + kx - 3;
            if (xs < 0 || xs >= Ww) continue;
            const size_t o = ((size_t)b * Hh + ys) * Ww + xs;
            s += lw[0][ky][kx] * avgp[o] + lw[1][ky][kx] * mxp[o];
        }
    }
    sa[((size_t)b * Hh + y) * Ww + xx] = 1.f / (1.f + __expf(-s));
}

// -------------------------------------------- horizontal 256-window sums ---
// One block per (b, y).  The 512-float row is DMA'd into LDS by the Tensor
// Data Mover (1D tile descriptor, built in SGPRs; lds_addr = 0 because row[]
// is the only LDS allocation in this kernel), then each thread emits one
// 256-wide window sum.
__global__ void hwin_sum(const float* __restrict__ sa, float* __restrict__ T) {
    __shared__ float row[Ww];            // at LDS offset 0 (sole allocation)
    const int y = blockIdx.x, b = blockIdx.y, tid = threadIdx.x;
    const unsigned long long gaddr =
        (unsigned long long)(const void*)(sa + ((size_t)b * Hh + y) * Ww);

    if (tid < 32) {                      // wave 0 issues the TDM transfer
        u32x4 g0;
        g0[0] = 1u;                                      // count=1, no gather
        g0[1] = 0u;                                      // lds_addr = 0
        g0[2] = (unsigned)(gaddr & 0xFFFFFFFFu);         // global_addr[31:0]
        g0[3] = (unsigned)(gaddr >> 32) | (2u << 30);    // addr[56:32] | type=2

        u32x8 g1;
        g1[0] = 2u << 16;                                // data_size = 4 bytes
        g1[1] = (Ww & 0xFFFFu) << 16;                    // tensor_dim0[15:0]
        g1[2] = (Ww >> 16) | (1u << 16);                 // dim0 hi | tensor_dim1=1
        g1[3] = ((unsigned)Ww) << 16;                    // tile_dim0 = 512
        g1[4] = 1u;                                      // tile_dim1=1, tile_dim2=0
        g1[5] = (unsigned)Ww;                            // tensor_dim0_stride lo
        g1[6] = 0u;                                      // stride hi / dim1_stride
        g1[7] = 0u;

        asm volatile("tensor_load_to_lds %0, %1"
                     :: "s"(g0), "s"(g1) : "memory");
        __builtin_amdgcn_s_wait_tensorcnt(0);            // s_wait_tensorcnt 0
    }
    __syncthreads();

    float s = 0.f;
    for (int i = 0; i < KCROP; ++i) s += row[tid + i];
    T[((size_t)b * Hh + y) * (Ww - KCROP) + tid] = s;
}

// ------------------------- vertical sliding window sums + global argmax ----
__global__ void vwin_argmax(const float* __restrict__ T, int* __restrict__ rowcol) {
    __shared__ float vals[256];
    __shared__ int   idxs[256];
    const int b = blockIdx.x, c = threadIdx.x;
    const float* Tb = T + (size_t)b * Hh * 256;

    float s = 0.f;
    for (int yy = 0; yy < KCROP; ++yy) s += Tb[(size_t)yy * 256 + c];
    float best = s; int bestIdx = c;                    // r = 0
    for (int r = 1; r < 256; ++r) {                     // slide the window
        s += Tb[(size_t)(r + 255) * 256 + c] - Tb[(size_t)(r - 1) * 256 + c];
        if (s > best) { best = s; bestIdx = r * 256 + c; }
    }
    vals[c] = best; idxs[c] = bestIdx;
    __syncthreads();
    for (int st = 128; st > 0; st >>= 1) {              // first-index tie-break
        if (c < st) {
            const float v2 = vals[c + st]; const int i2 = idxs[c + st];
            if (v2 > vals[c] || (v2 == vals[c] && i2 < idxs[c])) {
                vals[c] = v2; idxs[c] = i2;
            }
        }
        __syncthreads();
    }
    if (c == 0) { rowcol[b * 2] = idxs[0] >> 8; rowcol[b * 2 + 1] = idxs[0] & 255; }
}

// ----------------------------------------------------------------- crop ----
__global__ void crop_gather(const float* __restrict__ x, const int* __restrict__ rowcol,
                            float* __restrict__ out) {
    const int xx = threadIdx.x;
    const int yy = blockIdx.x;
    const int ci = blockIdx.y;
    const int b  = blockIdx.z;
    const int r  = rowcol[b * 2];
    const int cc = rowcol[b * 2 + 1];
    out[(((size_t)b * 3 + ci) * KCROP + yy) * KCROP + xx] =
        x[(((size_t)b * 3 + ci) * Hh + (r + yy)) * Ww + (cc + xx)];
}

// ---------------------------------------------------------------------------
extern "C" void kernel_launch(void* const* d_in, const int* in_sizes, int n_in,
                              void* d_out, int out_size, void* d_ws, size_t ws_size,
                              hipStream_t stream) {
    const float* x   = (const float*)d_in[0];
    const float* w1  = (const float*)d_in[1];
    const float* g1  = (const float*)d_in[2];
    const float* b1  = (const float*)d_in[3];
    const float* m1  = (const float*)d_in[4];
    const float* v1  = (const float*)d_in[5];
    const float* w2  = (const float*)d_in[6];
    const float* g2  = (const float*)d_in[7];
    const float* b2  = (const float*)d_in[8];
    const float* m2  = (const float*)d_in[9];
    const float* v2  = (const float*)d_in[10];
    const float* w3  = (const float*)d_in[11];
    const float* g3  = (const float*)d_in[12];
    const float* b3  = (const float*)d_in[13];
    const float* m3  = (const float*)d_in[14];
    const float* v3  = (const float*)d_in[15];
    const float* wsa = (const float*)d_in[16];

    // Workspace layout (fp16 activations halve HBM traffic for the big stages)
    char* ws = (char*)d_ws;
    const size_t HB = (size_t)Bb * C1 * Hh * Ww * sizeof(_Float16);  // 256 MB
    _Float16* h1   = (_Float16*)ws;
    _Float16* h2   = (_Float16*)(ws + HB);
    float*    avgp = (float*)(ws + 2 * HB);
    float*    mxp  = avgp + (size_t)Bb * Hh * Ww;
    float*    sa   = mxp  + (size_t)Bb * Hh * Ww;
    float*    T    = sa   + (size_t)Bb * Hh * Ww;
    int*      rc   = (int*)(T + (size_t)Bb * Hh * (Ww - KCROP));

    conv1_bn_relu<<<dim3(2, Hh, Bb * C1), 256, 0, stream>>>(x, w1, g1, b1, m1, v1, h1);

    conv_wmma<false><<<dim3(Bb * Hh * (Ww / TPX)), 256, 0, stream>>>(
        h1, w2, g2, b2, m2, v2, h2, nullptr, nullptr);

    conv_wmma<true><<<dim3(Bb * Hh * (Ww / TPX)), 256, 0, stream>>>(
        h2, w3, g3, b3, m3, v3, nullptr, avgp, mxp);

    sa_conv_sigmoid<<<dim3(Hh * Ww / 256, Bb), 256, 0, stream>>>(avgp, mxp, wsa, sa);

    hwin_sum<<<dim3(Hh, Bb), 256, 0, stream>>>(sa, T);

    vwin_argmax<<<dim3(Bb), 256, 0, stream>>>(T, rc);

    crop_gather<<<dim3(KCROP, 3, Bb), 256, 0, stream>>>(x, rc, (float*)d_out);
}
